// MapleSparseMoeBlock_2456721293596
// MI455X (gfx1250) — matrix operand order repaired and checked by reading, not compiled, not run on hardware
//
#include <hip/hip_runtime.h>
#include <hip/hip_bf16.h>
#include <math.h>

#define HID    2048
#define INTER  1408
#define NEXP   8
#define TOKENS 2048
#define NSLOTS (TOKENS * 2)

typedef __attribute__((ext_vector_type(16))) __bf16 v16bf;
typedef __attribute__((ext_vector_type(8)))  __bf16 v8bf;
typedef __attribute__((ext_vector_type(8)))  float  v8f;

// ---------------------------------------------------------------------------
// Init: zero the output accumulator and the per-expert slot counters.
// ---------------------------------------------------------------------------
__global__ __launch_bounds__(256) void init_kernel(float* __restrict__ out, int n,
                                                   int* __restrict__ counts) {
    int i = blockIdx.x * 256 + threadIdx.x;
    if (i < n) out[i] = 0.0f;
    if (blockIdx.x == 0 && threadIdx.x < NEXP) counts[threadIdx.x] = 0;
}

// ---------------------------------------------------------------------------
// Router: fp32 logits -> softmax -> top-2 -> renormalized weights.
// One block per token. Appends slot ids (token*2+k) to per-expert lists.
// ---------------------------------------------------------------------------
__global__ __launch_bounds__(256) void router_kernel(
    const float* __restrict__ x, const float* __restrict__ rw,
    int* __restrict__ counts, int* __restrict__ slot_list,
    float* __restrict__ w_slots) {
    const int t = blockIdx.x;
    const float* xr = x + (size_t)t * HID;

    float acc[NEXP];
#pragma unroll
    for (int e = 0; e < NEXP; ++e) acc[e] = 0.0f;
    for (int h = threadIdx.x; h < HID; h += 256) {
        const float xv = xr[h];
#pragma unroll
        for (int e = 0; e < NEXP; ++e) acc[e] += xv * rw[e * HID + h];
    }

    __shared__ float red[NEXP][256];
#pragma unroll
    for (int e = 0; e < NEXP; ++e) red[e][threadIdx.x] = acc[e];
    __syncthreads();

    __shared__ float logits[NEXP];
    if (threadIdx.x < NEXP) {
        float s = 0.0f;
        for (int i = 0; i < 256; ++i) s += red[threadIdx.x][i];
        logits[threadIdx.x] = s;
    }
    __syncthreads();

    if (threadIdx.x == 0) {
        float p[NEXP];
        float mx = logits[0];
        for (int e = 1; e < NEXP; ++e) mx = fmaxf(mx, logits[e]);
        float sum = 0.0f;
        for (int e = 0; e < NEXP; ++e) { p[e] = __expf(logits[e] - mx); sum += p[e]; }

        int e0 = 0;
        for (int e = 1; e < NEXP; ++e) if (p[e] > p[e0]) e0 = e;
        int e1 = (e0 == 0) ? 1 : 0;
        for (int e = 0; e < NEXP; ++e) if (e != e0 && p[e] > p[e1]) e1 = e;

        const float s0 = p[e0] / sum, s1 = p[e1] / sum;
        const float inv = 1.0f / (s0 + s1 + 1e-20f);

        int j0 = atomicAdd(&counts[e0], 1);
        slot_list[e0 * TOKENS + j0] = t * 2;
        w_slots[t * 2] = s0 * inv;

        int j1 = atomicAdd(&counts[e1], 1);
        slot_list[e1 * TOKENS + j1] = t * 2 + 1;
        w_slots[t * 2 + 1] = s1 * inv;
    }
}

// ---------------------------------------------------------------------------
// Grouped expert GEMM with bf16 WMMA (16x16x32), fp32 accumulation.
//   DOWN = false: C[slot, n] = x[tok]·W[e][n][:]   (gate & up fused via grid.x)
//   DOWN = true : out[tok, n] += w_slots[slot] * act[slot]·Wd[e][n][:]
// Block tile 128M x 128N, K step 32, LDS-staged bf16 (rows padded to 80 B).
// 8 waves: each computes 32M x 64N = 8 WMMA fragments per K step.
// ---------------------------------------------------------------------------
template <bool DOWN>
__global__ __launch_bounds__(256) void moe_gemm(
    const float* __restrict__ xa,   // A rows: x [T,H] or act [NSLOTS,I]
    const float* __restrict__ wA,   // gate weights or down weights [E][N][K]
    const float* __restrict__ wB,   // up weights (phase A only)
    float* __restrict__ oA,         // g_buf or out
    float* __restrict__ oB,         // u_buf (phase A only)
    const int* __restrict__ counts,
    const int* __restrict__ slot_list,
    const float* __restrict__ w_slots) {
    constexpr int KT = DOWN ? INTER : HID;  // reduction dim
    constexpr int NT = DOWN ? HID : INTER;  // output dim (per matrix)

    const int e = blockIdx.z;
    const int cnt = counts[e];
    const int m0 = blockIdx.y * 128;
    if (m0 >= cnt) return;  // uniform per block; EXEC stays all-ones for WMMA

    const float* W;
    float* OUT;
    int n0;
    if (DOWN) {
        W = wA; OUT = oA; n0 = blockIdx.x * 128;
    } else {
        const int ntx = blockIdx.x;
        if (ntx < (INTER / 128)) { W = wA; OUT = oA; n0 = ntx * 128; }
        else                     { W = wB; OUT = oB; n0 = (ntx - INTER / 128) * 128; }
    }

    const int tid = threadIdx.x;
    const int lane = tid & 31;
    const int wid = tid >> 5;
    const int wm = wid & 3;    // 0..3 -> 32-row strip
    const int wn = wid >> 2;   // 0..1 -> 64-col strip
    const int lh = lane >> 4;  // half-wave select
    const int l15 = lane & 15;

    __shared__ __bf16 As[128][40];  // [m][k], pitch 80 B (16 B aligned frags)
    __shared__ __bf16 Bs[128][40];  // [n][k]

    v8f c[2][4] = {};

    // Staging assignment: thread -> (row, k-half of 16 floats)
    const int ar = tid >> 1;
    const int ah = (tid & 1) * 16;

    const int* list = slot_list + e * TOKENS;

    const float* arow = nullptr;
    const bool avalid = (m0 + ar) < cnt;
    if (avalid) {
        const int slot = list[m0 + ar];
        const int row = DOWN ? slot : (slot >> 1);
        arow = xa + (size_t)row * KT;
    }
    const float* brow = W + (size_t)e * ((size_t)NT * KT) + (size_t)(n0 + ar) * KT;

    for (int k0 = 0; k0 < KT; k0 += 32) {
        {   // ---- stage A (fp32 -> bf16) ----
            __bf16* d = &As[ar][ah];
            if (avalid) {
                const float4* s = (const float4*)(arow + k0 + ah);
                float4 f0 = s[0], f1 = s[1], f2 = s[2], f3 = s[3];
                d[0]=(__bf16)f0.x; d[1]=(__bf16)f0.y; d[2]=(__bf16)f0.z; d[3]=(__bf16)f0.w;
                d[4]=(__bf16)f1.x; d[5]=(__bf16)f1.y; d[6]=(__bf16)f1.z; d[7]=(__bf16)f1.w;
                d[8]=(__bf16)f2.x; d[9]=(__bf16)f2.y; d[10]=(__bf16)f2.z; d[11]=(__bf16)f2.w;
                d[12]=(__bf16)f3.x; d[13]=(__bf16)f3.y; d[14]=(__bf16)f3.z; d[15]=(__bf16)f3.w;
            } else {
#pragma unroll
                for (int p = 0; p < 16; ++p) d[p] = (__bf16)0.0f;
            }
        }
        {   // ---- stage B (fp32 -> bf16) ----
            const float4* s = (const float4*)(brow + k0 + ah);
            float4 f0 = s[0], f1 = s[1], f2 = s[2], f3 = s[3];
            __bf16* d = &Bs[ar][ah];
            d[0]=(__bf16)f0.x; d[1]=(__bf16)f0.y; d[2]=(__bf16)f0.z; d[3]=(__bf16)f0.w;
            d[4]=(__bf16)f1.x; d[5]=(__bf16)f1.y; d[6]=(__bf16)f1.z; d[7]=(__bf16)f1.w;
            d[8]=(__bf16)f2.x; d[9]=(__bf16)f2.y; d[10]=(__bf16)f2.z; d[11]=(__bf16)f2.w;
            d[12]=(__bf16)f3.x; d[13]=(__bf16)f3.y; d[14]=(__bf16)f3.z; d[15]=(__bf16)f3.w;
        }
        __syncthreads();

        // ---- build fragments per ISA 7.12.2 layouts ----
        const int aoff = lh * 8;   // A: lanes>=16 hold K {8..15, 24..31}
        const int koff = lh * 8;   // B: lanes>=16 hold K {8..15, 24..31}
        v16bf a[2], b[4];
#pragma unroll
        for (int i = 0; i < 2; ++i) {
            const int row = wm * 32 + i * 16 + l15;
            v8bf lo = *(const v8bf*)&As[row][aoff];
            v8bf hi = *(const v8bf*)&As[row][aoff + 16];
            a[i] = __builtin_shufflevector(lo, hi, 0,1,2,3,4,5,6,7,8,9,10,11,12,13,14,15);
        }
#pragma unroll
        for (int j = 0; j < 4; ++j) {
            const int nrow = wn * 64 + j * 16 + l15;
            v8bf lo = *(const v8bf*)&Bs[nrow][koff];
            v8bf hi = *(const v8bf*)&Bs[nrow][koff + 16];
            b[j] = __builtin_shufflevector(lo, hi, 0,1,2,3,4,5,6,7,8,9,10,11,12,13,14,15);
        }
#pragma unroll
        for (int i = 0; i < 2; ++i)
#pragma unroll
            for (int j = 0; j < 4; ++j)
                c[i][j] = __builtin_amdgcn_wmma_f32_16x16x32_bf16(
                    false, a[i], false, b[j], (short)0, c[i][j], false, false);
        __syncthreads();
    }

    // ---- epilogue: C vgpr r holds M = r + 8*(lane>=16), N = lane&15 ----
#pragma unroll
    for (int i = 0; i < 2; ++i) {
#pragma unroll
        for (int j = 0; j < 4; ++j) {
            const int n = n0 + wn * 64 + j * 16 + l15;
#pragma unroll
            for (int r = 0; r < 8; ++r) {
                const int ml = wm * 32 + i * 16 + lh * 8 + r;
                const int jj = m0 + ml;
                if (jj < cnt) {
                    const int slot = list[jj];
                    const float v = c[i][j][r];
                    if (DOWN) {
                        const int tok = slot >> 1;
                        atomicAdd(OUT + (size_t)tok * NT + n, w_slots[slot] * v);
                    } else {
                        OUT[(size_t)slot * (size_t)NT + n] = v;
                    }
                }
            }
        }
    }
}

// ---------------------------------------------------------------------------
// SwiGLU: act = silu(g) * u, written in place over g (float4 vectorized).
// ---------------------------------------------------------------------------
__global__ __launch_bounds__(256) void silu_mul_kernel(float* __restrict__ g,
                                                       const float* __restrict__ u,
                                                       int n4) {
    const int i = blockIdx.x * 256 + threadIdx.x;
    if (i < n4) {
        float4 gv = ((const float4*)g)[i];
        float4 uv = ((const float4*)u)[i];
        float4 r;
        r.x = gv.x / (1.0f + __expf(-gv.x)) * uv.x;
        r.y = gv.y / (1.0f + __expf(-gv.y)) * uv.y;
        r.z = gv.z / (1.0f + __expf(-gv.z)) * uv.z;
        r.w = gv.w / (1.0f + __expf(-gv.w)) * uv.w;
        ((float4*)g)[i] = r;
    }
}

// ---------------------------------------------------------------------------
extern "C" void kernel_launch(void* const* d_in, const int* in_sizes, int n_in,
                              void* d_out, int out_size, void* d_ws, size_t ws_size,
                              hipStream_t stream) {
    (void)in_sizes; (void)n_in; (void)out_size; (void)ws_size;

    const float* x      = (const float*)d_in[0];  // [1, 2048, 2048]
    const float* rw     = (const float*)d_in[1];  // [8, 2048]
    const float* w_gate = (const float*)d_in[2];  // [8, 1408, 2048]
    const float* w_up   = (const float*)d_in[3];  // [8, 1408, 2048]
    const float* w_down = (const float*)d_in[4];  // [8, 2048, 1408]
    float* out = (float*)d_out;                   // [1, 2048, 2048]

    char* ws = (char*)d_ws;
    int*   counts    = (int*)(ws);                               // 32 B
    int*   slot_list = (int*)(ws + 1024);                        // 64 KB
    float* w_slots   = (float*)(ws + 1024 + NEXP * TOKENS * 4);  // 16 KB
    float* g_buf     = (float*)(ws + 131072);                    // 23 MB (act in place)
    float* u_buf     = g_buf + (size_t)NSLOTS * INTER;           // 23 MB

    init_kernel<<<(TOKENS * HID + 255) / 256, 256, 0, stream>>>(out, TOKENS * HID, counts);
    router_kernel<<<TOKENS, 256, 0, stream>>>(x, rw, counts, slot_list, w_slots);
    moe_gemm<false><<<dim3(2 * INTER / 128, TOKENS / 128, NEXP), 256, 0, stream>>>(
        x, w_gate, w_up, g_buf, u_buf, counts, slot_list, w_slots);
    silu_mul_kernel<<<((NSLOTS * INTER / 4) + 255) / 256, 256, 0, stream>>>(
        g_buf, u_buf, NSLOTS * INTER / 4);
    moe_gemm<true><<<dim3(HID / 128, TOKENS / 128, NEXP), 256, 0, stream>>>(
        g_buf, w_down, nullptr, out, nullptr, counts, slot_list, w_slots);
}
